// Dynamic_conv2d_49881750176387
// MI455X (gfx1250) — matrix operand order repaired and checked by reading, not compiled, not run on hardware
//
#include <hip/hip_runtime.h>
#include <hip/hip_fp16.h>

typedef __attribute__((ext_vector_type(16))) _Float16 v16h;
typedef __attribute__((ext_vector_type(8)))  _Float16 v8h;
typedef __attribute__((ext_vector_type(8)))  float    v8f;

#define B_  32
#define C_  128
#define O_  128
#define H_  56
#define W_  56
#define K_  4
#define HW_ (H_ * W_)
#define HP_ 58
#define WP_ 58
#define EPS_ 1e-5f

// ---------------- 1) global average pool: pooled[b*C+c] = mean(x[b,c,:,:]) ----
__global__ void pool_kernel(const float* __restrict__ x, float* __restrict__ pooled) {
    __shared__ float red[256];
    const int bc = blockIdx.x;                 // b*C + c
    const float* p = x + (size_t)bc * HW_;
    float s = 0.f;
    for (int i = threadIdx.x; i < HW_; i += 256) s += p[i];
    red[threadIdx.x] = s;
    __syncthreads();
    for (int off = 128; off > 0; off >>= 1) {
        if (threadIdx.x < off) red[threadIdx.x] += red[threadIdx.x + off];
        __syncthreads();
    }
    if (threadIdx.x == 0) pooled[bc] = red[0] * (1.f / (float)HW_);
}

// ---------------- 2) attention MLP + softmax + aggregated bias ---------------
__global__ void attn_kernel(const float* __restrict__ pooled,
                            const float* __restrict__ fc1_w, const float* __restrict__ fc1_b,
                            const float* __restrict__ fc2_w, const float* __restrict__ fc2_b,
                            const float* __restrict__ bias,
                            float* __restrict__ attn, float* __restrict__ aggb) {
    __shared__ float pl[C_];
    __shared__ float hk[K_];
    __shared__ float at[K_];
    const int b = blockIdx.x;
    const int tid = threadIdx.x;               // 128 threads
    pl[tid] = pooled[b * C_ + tid];
    __syncthreads();
    if (tid < K_) {
        float s = fc1_b[tid];
        for (int c = 0; c < C_; ++c) s += fc1_w[tid * C_ + c] * pl[c];
        hk[tid] = fmaxf(s, 0.f);
    }
    __syncthreads();
    if (tid < K_) {
        float s = fc2_b[tid];
        for (int k = 0; k < K_; ++k) s += fc2_w[tid * K_ + k] * hk[k];
        at[tid] = s;
    }
    __syncthreads();
    if (tid == 0) {
        float m = at[0];
        for (int k = 1; k < K_; ++k) m = fmaxf(m, at[k]);
        float e[K_], den = 0.f;
        for (int k = 0; k < K_; ++k) { e[k] = expf(at[k] - m); den += e[k]; }
        for (int k = 0; k < K_; ++k) { at[k] = e[k] / den; attn[b * K_ + k] = at[k]; }
    }
    __syncthreads();
    float s = 0.f;
    for (int k = 0; k < K_; ++k) s += at[k] * bias[k * O_ + tid];
    aggb[b * O_ + tid] = s;                    // tid == output channel
}

// ---------------- 3) zero-padded f16 NHWC copy of x: [B,58,58,C] -------------
// Channels innermost -> 3x3 patch rows are 16B-chunk contiguous in C for DMA.
__global__ void pad_kernel(const float* __restrict__ x, _Float16* __restrict__ xpad) {
    const int idx = blockIdx.x * 256 + threadIdx.x;   // < B*HP*WP*C (exact)
    const int c  = idx & (C_ - 1);
    const int ww = (idx >> 7) % WP_;
    const int hh = (idx / (C_ * WP_)) % HP_;
    const int b  = idx / (C_ * WP_ * HP_);
    float v = 0.f;
    if (hh >= 1 && hh <= H_ && ww >= 1 && ww <= W_)
        v = x[((size_t)(b * C_ + c) * H_ + (hh - 1)) * W_ + (ww - 1)];
    xpad[idx] = (_Float16)v;
}

// ---------------- 4) attention-blended weights, f16 [b][tap][O][C] -----------
__global__ void aggw_kernel(const float* __restrict__ weight, const float* __restrict__ attn,
                            _Float16* __restrict__ aggw) {
    const int idx = blockIdx.x * 256 + threadIdx.x;   // ((b*9+j)*128+o)*128+c
    const int c = idx & 127;
    const int o = (idx >> 7) & 127;
    const int j = (idx >> 14) % 9;
    const int b = idx / (16384 * 9);
    float s = 0.f;
#pragma unroll
    for (int k = 0; k < K_; ++k)
        s += attn[b * K_ + k] * weight[(((size_t)(k * O_ + o)) * C_ + c) * 9 + j];
    aggw[idx] = (_Float16)s;
}

// ------- helper: per-lane async DMA global(16B) -> LDS, tracked by ASYNCcnt --
__device__ __forceinline__ void async_b128(const _Float16* gptr, const _Float16* lptr) {
    const unsigned           lds  = (unsigned)(size_t)lptr;      // LDS byte offset
    const unsigned long long addr = (unsigned long long)(size_t)gptr;
    asm volatile("global_load_async_to_lds_b128 %0, %1, off"
                 :: "v"(lds), "v"(addr) : "memory");
}

// ---------------- 5) implicit-GEMM 3x3 conv via v_wmma_f32_16x16x32_f16 ------
// Block: 256 thr = 8 waves; sample b, M = 128 out-channels, N = 64 spatial.
// Wave w: rows [16w,16w+16) x four 16-col N-tiles.  K = 9 taps x 4 chunks of
// 32 input channels.  Both LDS operand tiles are filled by double-buffered
// global_load_async_to_lds_b128 DMA (3 issues/wave/step), synced with
// s_wait_asynccnt + barriers, and consumed by WMMA out of LDS.
__global__ void __launch_bounds__(256)
conv_wmma_kernel(const _Float16* __restrict__ xpad,   // NHWC padded
                 const _Float16* __restrict__ aggw,   // [b][tap][O][C]
                 const float* __restrict__ aggb,
                 float* __restrict__ out) {
    __shared__ _Float16 Wlds[2][128][40];  // [buf][o][k], stride 40 halves
    __shared__ _Float16 Plds[2][64][40];   // [buf][n][k]

    const int b    = blockIdx.y;
    const int n0   = blockIdx.x * 64;
    const int tid  = threadIdx.x;
    const int lane = tid & 31;
    const int wave = tid >> 5;
    const int m0   = wave * 16;

    v8f acc[4];
#pragma unroll
    for (int i = 0; i < 4; ++i)
#pragma unroll
        for (int r = 0; r < 8; ++r) acc[i][r] = 0.f;

    // ---- staging assignments (per thread, per K-step: 3 x b128 issues) ----
    // W tile: 128 rows x 32 halves = 512 x 16B  -> thread t: row t&127, parts {2*(t>>7), +1}
    const int wo = tid & 127;
    const int wp = (tid >> 7) * 2;             // 0 or 2 (16-byte parts)
    // P tile: 64 rows x 32 halves = 256 x 16B  -> thread t: row t>>2, part t&3
    const int pn = tid >> 2;                   // 0..63
    const int pp = tid & 3;                    // 0..3
    const int pg = n0 + pn;
    const int phh = pg / W_;                   // un-padded row of this n
    const int pww = pg % W_;
    const _Float16* xb = xpad + (size_t)b * (HP_ * WP_ * C_);

    // ---- fragment coords (wave32), contiguous K order shared by A and B ----
    //  A 16x32: lane&15 = M row; lanes<16 hold K 0..15, lanes>=16 hold K 16..31
    //  B 32x16: lane&15 = N col; lanes<16 hold K 0..15, lanes>=16 hold K 16..31
    const int arow  = m0 + (lane & 15);
    const int akoff = (lane < 16) ? 0 : 16;
    const int bkoff = (lane < 16) ? 0 : 16;

    auto stage = [&](int it, int buf) {
        const int j  = it >> 2;                // tap 0..8
        const int kc = it & 3;                 // K-chunk of 32 channels
        const int kh = j / 3, kw = j % 3;
        // weights: aggw[b][j][o][kc*32 ..]
        const _Float16* gw = aggw + ((size_t)(b * 9 + j) << 14) + wo * C_ + kc * 32;
        async_b128(gw + wp * 8,       &Wlds[buf][wo][wp * 8]);
        async_b128(gw + (wp + 1) * 8, &Wlds[buf][wo][(wp + 1) * 8]);
        // patch: xpad[b][phh+kh][pww+kw][kc*32 ..]  (NHWC, contiguous in C)
        const _Float16* gp = xb + ((size_t)(phh + kh) * WP_ + (pww + kw)) * C_
                           + kc * 32 + pp * 8;
        async_b128(gp, &Plds[buf][pn][pp * 8]);
    };

    stage(0, 0);                               // prologue fill of buffer 0

    for (int it = 0; it < 36; ++it) {
        const int buf = it & 1;
        __syncthreads();                       // everyone done READING buf^1
        if (it + 1 < 36) {
            stage(it + 1, buf ^ 1);            // prefetch next K-step
            asm volatile("s_wait_asynccnt 3" ::: "memory");  // current buf resident
        } else {
            asm volatile("s_wait_asynccnt 0" ::: "memory");
        }
        __syncthreads();                       // buf resident for all waves

        v8h alo = *(const v8h*)&Wlds[buf][arow][akoff];
        v8h ahi = *(const v8h*)&Wlds[buf][arow][akoff + 8];
        v16h af = __builtin_shufflevector(alo, ahi,
            0,1,2,3,4,5,6,7,8,9,10,11,12,13,14,15);
#pragma unroll
        for (int nt = 0; nt < 4; ++nt) {
            const int col = nt * 16 + (lane & 15);
            v8h blo = *(const v8h*)&Plds[buf][col][bkoff];
            v8h bhi = *(const v8h*)&Plds[buf][col][bkoff + 8];
            v16h bf = __builtin_shufflevector(blo, bhi,
                0,1,2,3,4,5,6,7,8,9,10,11,12,13,14,15);
            acc[nt] = __builtin_amdgcn_wmma_f32_16x16x32_f16(
                false, af, false, bf, (short)0, acc[nt], false, false);
        }
    }

    // Epilogue: C/D layout -> VGPR r holds M=r (lanes<16) / M=r+8 (lanes>=16)
    const int obase = m0 + ((lane < 16) ? 0 : 8);
#pragma unroll
    for (int r = 0; r < 8; ++r) {
        const int o = obase + r;
        const float ab = aggb[b * O_ + o];
        float* po = out + (size_t)(b * O_ + o) * HW_ + n0 + (lane & 15);
#pragma unroll
        for (int nt = 0; nt < 4; ++nt)
            po[nt * 16] = acc[nt][r] + ab;
    }
}

// ---------------- 6) BN stats, deterministic two-stage (no float atomics) ----
__global__ void stats_partial_kernel(const float* __restrict__ out, float* __restrict__ partial) {
    __shared__ float rs[256], rq[256];
    const int o = blockIdx.y;
    const int z = blockIdx.x;                  // 49 chunks x 2048 elems = 100352
    float s = 0.f, q = 0.f;
#pragma unroll
    for (int i = 0; i < 8; ++i) {
        const int g  = z * 2048 + i * 256 + threadIdx.x;
        const int bb = g / HW_;
        const int nn = g - bb * HW_;
        const float v = out[(size_t)(bb * O_ + o) * HW_ + nn];
        s += v; q += v * v;
    }
    rs[threadIdx.x] = s; rq[threadIdx.x] = q;
    __syncthreads();
    for (int off = 128; off > 0; off >>= 1) {
        if (threadIdx.x < off) {
            rs[threadIdx.x] += rs[threadIdx.x + off];
            rq[threadIdx.x] += rq[threadIdx.x + off];
        }
        __syncthreads();
    }
    if (threadIdx.x == 0) {
        partial[o * 49 + z]           = rs[0];
        partial[O_ * 49 + o * 49 + z] = rq[0];
    }
}

__global__ void stats_finalize_kernel(const float* __restrict__ partial, float* __restrict__ stats) {
    const int o = threadIdx.x;                 // 128 threads, 1 block
    float s = 0.f, q = 0.f;
    for (int z = 0; z < 49; ++z) {
        s += partial[o * 49 + z];
        q += partial[O_ * 49 + o * 49 + z];
    }
    stats[o]      = s;
    stats[O_ + o] = q;
}

// ---------------- 7) BN normalize + affine + ReLU, in place ------------------
__global__ void bn_relu_kernel(float* __restrict__ out, const float* __restrict__ stats,
                               const float* __restrict__ gamma, const float* __restrict__ beta) {
    const size_t idx = (size_t)blockIdx.x * 256 + threadIdx.x;
    const int o = (int)((idx / HW_) % O_);
    const float cnt  = (float)(B_ * HW_);
    const float mean = stats[o] / cnt;
    const float var  = stats[O_ + o] / cnt - mean * mean;
    const float inv  = rsqrtf(var + EPS_);
    const float v = out[idx];
    const float y = (v - mean) * inv * gamma[o] + beta[o];
    out[idx] = fmaxf(y, 0.f);
}

// ---------------- launch -----------------------------------------------------
extern "C" void kernel_launch(void* const* d_in, const int* in_sizes, int n_in,
                              void* d_out, int out_size, void* d_ws, size_t ws_size,
                              hipStream_t stream) {
    const float* x      = (const float*)d_in[0];
    const float* fc1_w  = (const float*)d_in[1];
    const float* fc1_b  = (const float*)d_in[2];
    const float* fc2_w  = (const float*)d_in[3];
    const float* fc2_b  = (const float*)d_in[4];
    const float* weight = (const float*)d_in[5];
    const float* bias   = (const float*)d_in[6];
    const float* gamma  = (const float*)d_in[7];
    const float* beta   = (const float*)d_in[8];
    float* out = (float*)d_out;

    // workspace layout (bytes)
    char* ws = (char*)d_ws;
    float*    pooled  = (float*)(ws + 0);                         // 16384
    float*    attn    = (float*)(ws + 16384);                     // 512
    float*    aggb    = (float*)(ws + 16896);                     // 16384
    float*    stats   = (float*)(ws + 33280);                     // 1024
    float*    partial = (float*)(ws + 34304);                     // 50176
    _Float16* aggw    = (_Float16*)(ws + 84992);                  // 9437184
    _Float16* xpad    = (_Float16*)(ws + 84992 + 9437184);        // 27557888 (NHWC)
    (void)ws_size; (void)in_sizes; (void)n_in; (void)out_size;

    pool_kernel<<<B_ * C_, 256, 0, stream>>>(x, pooled);
    attn_kernel<<<B_, 128, 0, stream>>>(pooled, fc1_w, fc1_b, fc2_w, fc2_b, bias, attn, aggb);
    pad_kernel<<<(B_ * HP_ * WP_ * C_) / 256, 256, 0, stream>>>(x, xpad);
    aggw_kernel<<<(B_ * 9 * O_ * C_) / 256, 256, 0, stream>>>(weight, attn, aggw);
    conv_wmma_kernel<<<dim3(HW_ / 64, B_), 256, 0, stream>>>(xpad, aggw, aggb, out);
    stats_partial_kernel<<<dim3(49, O_), 256, 0, stream>>>(out, partial);
    stats_finalize_kernel<<<1, O_, 0, stream>>>(partial, stats);
    bn_relu_kernel<<<(B_ * O_ * HW_) / 256, 256, 0, stream>>>(out, stats, gamma, beta);
}
